// Attention_670014898316
// MI455X (gfx1250) — compile-verified, matrix operand order checked by reading, not run on hardware
//
#include <hip/hip_runtime.h>
#include <hip/hip_bf16.h>
#include <stdint.h>

// Shapes are fixed by the reference: B=1, S=4096, E=768, H=12, D=64.
#define S_LEN 4096
#define EMB   768
#define NHEAD 12
#define HDIM  64

typedef __attribute__((ext_vector_type(16))) __bf16 v16bf;
typedef __attribute__((ext_vector_type(8)))  float  v8f;

union AFrag { v16bf v; unsigned int u[8]; };

// ---------------------------------------------------------------------------
// gfx1250 async global->LDS path (ASYNCcnt), guarded so compile never breaks.
// Probe round 2 confirmed the builtin exists with param0 =
//   'int __attribute__((vector_size(16))) __device__ *'  (AS1 int4*),
// so: (global int4* src, lds int4* dst, imm offset, imm cpol).
// ---------------------------------------------------------------------------
#if defined(__gfx1250__) && \
    __has_builtin(__builtin_amdgcn_global_load_async_to_lds_b128) && \
    __has_builtin(__builtin_amdgcn_s_wait_asynccnt)
#define HAVE_ASYNC_LDS 1
#else
#define HAVE_ASYNC_LDS 0
#endif

typedef int gvec4i __attribute__((vector_size(16)));
typedef gvec4i __attribute__((address_space(1))) as1_v4i;
typedef gvec4i __attribute__((address_space(3))) as3_v4i;

// Copy 16B global -> LDS. Async (no VGPR data, ASYNCcnt-tracked) when possible.
__device__ __forceinline__ void cp_b128(__bf16* lds_dst, const __bf16* gsrc) {
#if HAVE_ASYNC_LDS
  __builtin_amdgcn_global_load_async_to_lds_b128(
      (as1_v4i*)gsrc, (as3_v4i*)lds_dst, /*offset=*/0, /*cpol=*/0);
#else
  *(uint4*)lds_dst = *(const uint4*)gsrc;
#endif
}

template <int N>
__device__ __forceinline__ void async_wait_le() {
#if HAVE_ASYNC_LDS
  __builtin_amdgcn_s_wait_asynccnt((short)N);
#endif
}

__device__ __forceinline__ __bf16 f2bf(float f) {
  union { float f; unsigned u; } x; x.f = f;
  unsigned r = (x.u + 0x7fffu + ((x.u >> 16) & 1u)) >> 16;  // RNE
  unsigned short h = (unsigned short)r;
  union { unsigned short s; __bf16 b; } y; y.s = h;
  return y.b;
}

// Load a 16x32 bf16 WMMA operand fragment from LDS.
// Layout per CDNA5 ISA 7.12.2 (16-bit A, 16x32): lane L holds row M=L&15;
// VGPR r holds packed K pair k=(r<4 ? 2r : 2r+8) + (L>=16 ? 8 : 0).
// B operand is supplied as B^T rows using the same layout (lane carries N).
__device__ __forceinline__ v16bf lds_frag16x32(const __bf16* base, int stride, int koff) {
  const int lane = threadIdx.x & 31;
  const int row  = lane & 15;
  const int khi  = (lane >> 4) << 3;
  const unsigned short* p = (const unsigned short*)(base + row * stride + koff);
  AFrag f;
#pragma unroll
  for (int r = 0; r < 8; ++r) {
    const int k = ((r < 4) ? (2 * r) : (2 * r + 8)) + khi;
    f.u[r] = *(const unsigned int*)(p + k);  // (k,k+1) pair, 4B aligned
  }
  return f.v;
}

__device__ __forceinline__ v8f wmma_bf16(v16bf a, v16bf b, v8f c) {
  return __builtin_amdgcn_wmma_f32_16x16x32_bf16(
      /*neg_a=*/false, a, /*neg_b=*/false, b,
      /*c_mod=*/(short)0, c, /*reuse_a=*/false, /*reuse_b=*/false);
}

// ---------------------------------------------------------------------------
// f32 -> bf16 conversion (used once for the four weight matrices)
// ---------------------------------------------------------------------------
__global__ __launch_bounds__(256) void cvt_f32_bf16(const float* __restrict__ in,
                                                    __bf16* __restrict__ out, int n) {
  int i = (blockIdx.x * 256 + threadIdx.x) * 4;
  if (i + 3 < n) {
    float4 v = *(const float4*)(in + i);
    out[i + 0] = f2bf(v.x); out[i + 1] = f2bf(v.y);
    out[i + 2] = f2bf(v.z); out[i + 3] = f2bf(v.w);
  }
}

// ---------------------------------------------------------------------------
// O = X(MxK) @ W^T  with W stored [N][K] (nn.Linear), bf16 WMMA, f32 accum.
//   TX = float  : activations converted to bf16 while staging into LDS
//   TX = __bf16 : already-converted input (ctx), staged via async-to-LDS
//   MODE 0: bf16 output in head-split layout [head][S][64], scaled
//   MODE 1: f32  output row-major [M][N]
// Block tile 128x64, 256 threads (8 wave32), wave tile 32x32 (2x2 WMMA accs).
// K loop is double-buffered: tile t+1 streams into LDS (async when available)
// while tile t feeds the WMMA pipe; fence = s_wait_asynccnt <= in-flight.
// ---------------------------------------------------------------------------
template <typename TX, int MODE>
__global__ __launch_bounds__(256) void gemm_xwt(const TX* __restrict__ X,
                                                const __bf16* __restrict__ W,
                                                void* __restrict__ out,
                                                float scale) {
  constexpr int K = EMB, N = EMB;
  constexpr int KTILES = K / 32;
  // per-thread async copies issued per tile stage (W slab always async-capable)
  constexpr int ASYNC_PER_TILE = (sizeof(TX) == 2) ? 3 : 1;

  __shared__ __bf16 Xs[2][128 * 40];  // 128x32 tiles, stride 40 (bank-spread)
  __shared__ __bf16 Ws[2][64 * 40];   // 64x32 tiles

  const int m0 = blockIdx.x * 128;
  const int n0 = blockIdx.y * 64;
  const int tid  = threadIdx.x;
  const int w    = tid >> 5;
  const int lane = tid & 31;
  const int mrow = (w >> 1) * 32;   // wave's M offset inside tile
  const int ncol = (w & 1) * 32;    // wave's N offset inside tile

  v8f acc[2][2] = {};

  const int xrow = tid >> 1, xoff = (tid & 1) * 16;  // 128 rows, 16 elems/thread
  const int wrow = tid >> 2, woff = (tid & 3) * 8;   // 64 rows,  8 elems/thread

  auto stage = [&](int k0, int b) {
    if constexpr (sizeof(TX) == 4) {
      const float* gx = (const float*)X + (size_t)(m0 + xrow) * K + k0 + xoff;
      float4 v0 = *(const float4*)(gx + 0);
      float4 v1 = *(const float4*)(gx + 4);
      float4 v2 = *(const float4*)(gx + 8);
      float4 v3 = *(const float4*)(gx + 12);
      __bf16* d = &Xs[b][xrow * 40 + xoff];
      d[0]  = f2bf(v0.x); d[1]  = f2bf(v0.y); d[2]  = f2bf(v0.z); d[3]  = f2bf(v0.w);
      d[4]  = f2bf(v1.x); d[5]  = f2bf(v1.y); d[6]  = f2bf(v1.z); d[7]  = f2bf(v1.w);
      d[8]  = f2bf(v2.x); d[9]  = f2bf(v2.y); d[10] = f2bf(v2.z); d[11] = f2bf(v2.w);
      d[12] = f2bf(v3.x); d[13] = f2bf(v3.y); d[14] = f2bf(v3.z); d[15] = f2bf(v3.w);
      if (k0 + 32 < K) __builtin_prefetch(gx + 32, 0, 1);
    } else {
      const __bf16* gx = (const __bf16*)X + (size_t)(m0 + xrow) * K + k0 + xoff;
      cp_b128(&Xs[b][xrow * 40 + xoff], gx);
      cp_b128(&Xs[b][xrow * 40 + xoff + 8], gx + 8);
    }
    const __bf16* gw = W + (size_t)(n0 + wrow) * K + k0 + woff;
    cp_b128(&Ws[b][wrow * 40 + woff], gw);
  };

  stage(0, 0);  // prologue: tile 0 in flight

  for (int it = 0; it < KTILES; ++it) {
    const int b = it & 1;
    if (it + 1 < KTILES) stage((it + 1) * 32, b ^ 1);  // issue next tile
    async_wait_le<ASYNC_PER_TILE>();  // tile b landed; only next-tile in flight
    __syncthreads();

    v16bf a0 = lds_frag16x32(&Xs[b][(mrow)      * 40], 40, 0);
    v16bf a1 = lds_frag16x32(&Xs[b][(mrow + 16) * 40], 40, 0);
    v16bf b0 = lds_frag16x32(&Ws[b][(ncol)      * 40], 40, 0);
    v16bf b1 = lds_frag16x32(&Ws[b][(ncol + 16) * 40], 40, 0);
    acc[0][0] = wmma_bf16(a0, b0, acc[0][0]);
    acc[0][1] = wmma_bf16(a0, b1, acc[0][1]);
    acc[1][0] = wmma_bf16(a1, b0, acc[1][0]);
    acc[1][1] = wmma_bf16(a1, b1, acc[1][1]);
    __syncthreads();  // everyone done with buffer b before it is restaged
  }

  // ---- epilogue: C/D layout lane->(n = lane&15, m = r + 8*(lane>=16)) ----
  const int nlo = lane & 15;
  const int mhi = (lane >> 4) << 3;
#pragma unroll
  for (int i = 0; i < 2; ++i)
#pragma unroll
    for (int j = 0; j < 2; ++j) {
#pragma unroll
      for (int r = 0; r < 8; ++r) {
        const int m = m0 + mrow + i * 16 + r + mhi;
        const int n = n0 + ncol + j * 16 + nlo;
        const float vv = acc[i][j][r] * scale;
        if constexpr (MODE == 0) {
          __bf16* ob = (__bf16*)out;
          const int head = n >> 6, col = n & 63;
          ob[(size_t)head * S_LEN * HDIM + (size_t)m * HDIM + col] = f2bf(vv);
        } else {
          ((float*)out)[(size_t)m * N + n] = vv;
        }
      }
    }
}

// ---------------------------------------------------------------------------
// Flash attention: one block = one head x 64 query rows.
// Online softmax over 64-wide key blocks; ctx accumulators live in registers
// (each wave owns two 16x16 output tiles across the whole key loop).
// Q/K tiles stream in via async global->LDS; V is transposed while staging.
// ---------------------------------------------------------------------------
__global__ __launch_bounds__(256) void flash_attn(const __bf16* __restrict__ Qh,
                                                  const __bf16* __restrict__ Kh,
                                                  const __bf16* __restrict__ Vh,
                                                  __bf16* __restrict__ ctx) {
  __shared__ __bf16 Qs[64 * 72];
  __shared__ __bf16 Ks[64 * 72];
  __shared__ __bf16 Vt[64 * 72];   // V transposed: [d][key]
  __shared__ float  Sc[64 * 64];   // f32 scores
  __shared__ __bf16 Ps[64 * 72];   // softmax probs (bf16)
  __shared__ float  rowM[64], rowL[64], rowA[64];

  const int q0   = blockIdx.x * 64;
  const int h    = blockIdx.y;
  const int tid  = threadIdx.x;
  const int w    = tid >> 5;
  const int lane = tid & 31;
  const int mt   = w >> 1;          // wave's row-tile (0..3)
  const int ntb  = (w & 1) * 2;     // wave's first col-tile (0 or 2)
  const int nlo  = lane & 15;
  const int mhi  = (lane >> 4) << 3;

  const size_t hb = (size_t)h * S_LEN * HDIM;
  const int row = tid >> 2, c0 = (tid & 3) * 16;

  {  // load Q tile (Q already carries the 1/sqrt(D) scale)
    const __bf16* gq = Qh + hb + (size_t)(q0 + row) * HDIM + c0;
    cp_b128(&Qs[row * 72 + c0], gq);
    cp_b128(&Qs[row * 72 + c0 + 8], gq + 8);
  }
  if (tid < 64) { rowM[tid] = -3.0e38f; rowL[tid] = 0.0f; }
  v8f acc0 = {}, acc1 = {};
  async_wait_le<0>();
  __syncthreads();

  for (int kb = 0; kb < S_LEN / 64; ++kb) {
    const int k0 = kb * 64;
    {  // K tile async row-major; V tile transposed into Vt
      const __bf16* gk = Kh + hb + (size_t)(k0 + row) * HDIM + c0;
      cp_b128(&Ks[row * 72 + c0], gk);
      cp_b128(&Ks[row * 72 + c0 + 8], gk + 8);
      const __bf16* gv = Vh + hb + (size_t)(k0 + row) * HDIM + c0;
      __bf16 tmp[16];
      *(uint4*)&tmp[0] = *(const uint4*)gv;
      *(uint4*)&tmp[8] = *(const uint4*)(gv + 8);
#pragma unroll
      for (int j = 0; j < 16; ++j) Vt[(c0 + j) * 72 + row] = tmp[j];
    }
    async_wait_le<0>();
    __syncthreads();

    {  // scores = Q @ K^T  (two 16x16 tiles per wave, K-dim = 64 = 2 steps)
      v16bf aq0 = lds_frag16x32(&Qs[mt * 16 * 72], 72, 0);
      v16bf aq1 = lds_frag16x32(&Qs[mt * 16 * 72], 72, 32);
#pragma unroll
      for (int j = 0; j < 2; ++j) {
        const int nt = ntb + j;
        v16bf bk0 = lds_frag16x32(&Ks[nt * 16 * 72], 72, 0);
        v16bf bk1 = lds_frag16x32(&Ks[nt * 16 * 72], 72, 32);
        v8f s = {};
        s = wmma_bf16(aq0, bk0, s);
        s = wmma_bf16(aq1, bk1, s);
#pragma unroll
        for (int r = 0; r < 8; ++r)
          Sc[(mt * 16 + r + mhi) * 64 + nt * 16 + nlo] = s[r];
      }
    }
    __syncthreads();

    if (tid < 64) {  // online softmax, one thread per query row
      const float mold = rowM[tid];
      float mloc = mold;
#pragma unroll 8
      for (int j = 0; j < 64; ++j) mloc = fmaxf(mloc, Sc[tid * 64 + j]);
      const float alpha = __expf(mold - mloc);
      float sum = 0.0f;
#pragma unroll 8
      for (int j = 0; j < 64; ++j) {
        const float p = __expf(Sc[tid * 64 + j] - mloc);
        sum += p;
        Ps[tid * 72 + j] = f2bf(p);
      }
      rowL[tid] = rowL[tid] * alpha + sum;
      rowM[tid] = mloc;
      rowA[tid] = alpha;
    }
    __syncthreads();

    {  // rescale running accumulators, then ctx += P @ V
#pragma unroll
      for (int r = 0; r < 8; ++r) {
        const float a = rowA[mt * 16 + r + mhi];
        acc0[r] *= a;
        acc1[r] *= a;
      }
      v16bf ap0 = lds_frag16x32(&Ps[mt * 16 * 72], 72, 0);
      v16bf ap1 = lds_frag16x32(&Ps[mt * 16 * 72], 72, 32);
      v16bf bv;
      bv = lds_frag16x32(&Vt[(ntb)     * 16 * 72], 72, 0);  acc0 = wmma_bf16(ap0, bv, acc0);
      bv = lds_frag16x32(&Vt[(ntb)     * 16 * 72], 72, 32); acc0 = wmma_bf16(ap1, bv, acc0);
      bv = lds_frag16x32(&Vt[(ntb + 1) * 16 * 72], 72, 0);  acc1 = wmma_bf16(ap0, bv, acc1);
      bv = lds_frag16x32(&Vt[(ntb + 1) * 16 * 72], 72, 32); acc1 = wmma_bf16(ap1, bv, acc1);
    }
    __syncthreads();
  }

  // epilogue: ctx[s][h*64 + d] = acc / l
#pragma unroll
  for (int r = 0; r < 8; ++r) {
    const int m = mt * 16 + r + mhi;
    const float linv = 1.0f / rowL[m];
    const size_t orow = (size_t)(q0 + m) * EMB + h * HDIM;
    ctx[orow + (ntb)     * 16 + nlo] = f2bf(acc0[r] * linv);
    ctx[orow + (ntb + 1) * 16 + nlo] = f2bf(acc1[r] * linv);
  }
}

// ---------------------------------------------------------------------------
extern "C" void kernel_launch(void* const* d_in, const int* in_sizes, int n_in,
                              void* d_out, int out_size, void* d_ws, size_t ws_size,
                              hipStream_t stream) {
  const float* Xq = (const float*)d_in[0];
  const float* Xk = (const float*)d_in[1];
  const float* Xv = (const float*)d_in[2];
  const float* Wq = (const float*)d_in[3];
  const float* Wk = (const float*)d_in[4];
  const float* Wv = (const float*)d_in[5];
  const float* Wo = (const float*)d_in[6];

  const size_t SE = (size_t)S_LEN * EMB;   // 3,145,728
  const size_t EE = (size_t)EMB * EMB;     //   589,824

  // workspace layout (bf16): 4 weights + Q/K/V head-split + ctx  (~30 MB)
  __bf16* Wqb = (__bf16*)d_ws;
  __bf16* Wkb = Wqb + EE;
  __bf16* Wvb = Wkb + EE;
  __bf16* Wob = Wvb + EE;
  __bf16* Qh  = Wob + EE;
  __bf16* Kh  = Qh + SE;
  __bf16* Vh  = Kh + SE;
  __bf16* ctx = Vh + SE;

  // 1) weights -> bf16 (once per call; activations convert inside the GEMM)
  cvt_f32_bf16<<<(int)(EE / 1024), 256, 0, stream>>>(Wq, Wqb, (int)EE);
  cvt_f32_bf16<<<(int)(EE / 1024), 256, 0, stream>>>(Wk, Wkb, (int)EE);
  cvt_f32_bf16<<<(int)(EE / 1024), 256, 0, stream>>>(Wv, Wvb, (int)EE);
  cvt_f32_bf16<<<(int)(EE / 1024), 256, 0, stream>>>(Wo, Wob, (int)EE);

  // 2) Q/K/V projections (Q carries the 1/sqrt(64) softmax scale)
  dim3 gg(S_LEN / 128, EMB / 64);  // (32, 12)
  gemm_xwt<float, 0><<<gg, 256, 0, stream>>>(Xq, Wqb, Qh, 0.125f);
  gemm_xwt<float, 0><<<gg, 256, 0, stream>>>(Xk, Wkb, Kh, 1.0f);
  gemm_xwt<float, 0><<<gg, 256, 0, stream>>>(Xv, Wvb, Vh, 1.0f);

  // 3) flash attention: grid = (64 q-blocks, 12 heads)
  flash_attn<<<dim3(S_LEN / 64, NHEAD), 256, 0, stream>>>(Qh, Kh, Vh, ctx);

  // 4) output projection -> f32 d_out [S][E]
  gemm_xwt<__bf16, 1><<<gg, 256, 0, stream>>>(ctx, Wob, d_out, 1.0f);
}
